// S6Block_38963943309328
// MI455X (gfx1250) — matrix operand order repaired and checked by reading, not compiled, not run on hardware
//
#include <hip/hip_runtime.h>

typedef __bf16 bf16_t;
typedef __attribute__((ext_vector_type(16))) __bf16 v16bf;
typedef __attribute__((ext_vector_type(8)))  float  v8f;
typedef __attribute__((ext_vector_type(4)))  int    v4i;

#define D_MODEL 1024
#define D_STATE 64
#define D_INNER 2048
#define BATCH   4
#define SEQL    2048
#define ROWS    (BATCH * SEQL)   // 8192

#define GLOBAL_AS __attribute__((address_space(1)))
#define LOCAL_AS  __attribute__((address_space(3)))

// ---- gfx1250 async-to-LDS path (probe-confirmed present on this toolchain) ----
#if defined(__has_builtin)
#if __has_builtin(__builtin_amdgcn_global_load_async_to_lds_b128)
#define HAVE_ASYNC_LDS 1
#else
#define HAVE_ASYNC_LDS 0
#endif
#if __has_builtin(__builtin_amdgcn_s_wait_asynccnt)
#define HAVE_WAIT_ASYNC 1
#else
#define HAVE_WAIT_ASYNC 0
#endif
#else
#define HAVE_ASYNC_LDS 0
#define HAVE_WAIT_ASYNC 0
#endif

__device__ __forceinline__ void async_copy_b128(const bf16_t* g, bf16_t* l) {
#if HAVE_ASYNC_LDS
  __builtin_amdgcn_global_load_async_to_lds_b128(
      (GLOBAL_AS v4i*)g, (LOCAL_AS v4i*)l, 0, 0);
#else
  *(uint4*)l = *(const uint4*)g;  // fallback (never used when builtin present)
#endif
}

__device__ __forceinline__ void wait_async_zero() {
#if HAVE_WAIT_ASYNC
  __builtin_amdgcn_s_wait_asynccnt(0);
#else
  asm volatile("s_wait_asynccnt 0x0" ::: "memory");
#endif
}

__device__ __forceinline__ float siluf(float v) {
  return v * __builtin_amdgcn_rcpf(1.0f + __expf(-v));
}
// stable + cheap: max(v,0) + log(1+exp(-|v|))  -> v_exp_f32 + v_log_f32
__device__ __forceinline__ float softplusf(float v) {
  return fmaxf(v, 0.0f) + __logf(1.0f + __expf(-fabsf(v)));
}

// ---------------------------------------------------------------------------
// fp32 -> bf16 conversion (compact, same layout)
// ---------------------------------------------------------------------------
__global__ void cvt_to_bf16(const float* __restrict__ src, bf16_t* __restrict__ dst,
                            long total) {
  long stride = (long)gridDim.x * blockDim.x;
  for (long i = (long)blockIdx.x * blockDim.x + threadIdx.x; i < total; i += stride)
    dst[i] = (bf16_t)src[i];
}

// ---------------------------------------------------------------------------
// fp32 (K x N) -> bf16 transposed (N x K), LDS-tiled, coalesced both ways
// ---------------------------------------------------------------------------
__global__ __launch_bounds__(256) void cvt_transpose_bf16(
    const float* __restrict__ src, bf16_t* __restrict__ dst, int K, int N) {
  __shared__ float tile[32][33];
  const int k0 = blockIdx.y << 5;
  const int n0 = blockIdx.x << 5;
  const int tx = threadIdx.x & 31;
  const int ty = threadIdx.x >> 5;   // 0..7
  #pragma unroll
  for (int i = 0; i < 32; i += 8)
    tile[ty + i][tx] = src[(size_t)(k0 + ty + i) * N + (n0 + tx)];
  __syncthreads();
  #pragma unroll
  for (int i = 0; i < 32; i += 8)
    dst[(size_t)(n0 + ty + i) * K + (k0 + tx)] = (bf16_t)tile[tx][ty + i];
}

// ---------------------------------------------------------------------------
// bf16 WMMA GEMM: out[M,N] = act(A[M,K] @ W[K,N] + bias) (+ resid)
// Wt is the PRE-TRANSPOSED weight, N x K row-major (K contiguous).
// block = 256 threads (8 waves); block tile 64x128; wave tile 32x32 (2x2 WMMA)
// K-step 64, double-buffered LDS, ONE barrier per K-iteration.
// Prefetch via GLOBAL_LOAD_ASYNC_TO_LDS_B128 (ASYNCcnt): memory->LDS without
// touching VGPRs, so no staging registers and no scratch spills.
// act: 0 = none, 1 = silu, 2 = softplus
// Optional bf16 duplicate output for cols < bfW (fused down-conversion).
// M%64==0, N%128==0, K%64==0 (true for all call sites).
// ---------------------------------------------------------------------------
__global__ __launch_bounds__(256, 1) void gemm_bf16_wmma(
    const bf16_t* __restrict__ A, const bf16_t* __restrict__ Wt,
    const float* __restrict__ bias, float* __restrict__ out,
    const float* __restrict__ resid, bf16_t* __restrict__ bfout, int bfW,
    int M, int N, int K, int act) {
  __shared__ __attribute__((aligned(16))) bf16_t As[2][64][72];    // [m][k]
  __shared__ __attribute__((aligned(16))) bf16_t Bs[2][128][72];   // [n][k]

  const int t    = threadIdx.x;
  const int lane = t & 31;
  const int wave = t >> 5;
  const int wy   = wave >> 2;   // 0..1
  const int wx   = wave & 3;    // 0..3
  const int lr   = lane & 15;
  const int hi   = lane >> 4;   // 0 or 1

  const int rowBase = blockIdx.y * 64;
  const int colBase = blockIdx.x * 128;

  // staging map: A tile 64x64 -> 2x b128/thread; B tile 128x64 -> 4x b128/thread
  const int aR = t >> 2;            // 0..63
  const int aC = (t & 3) << 4;      // 0,16,32,48
  const int bR = t >> 1;            // 0..127
  const int bC = (t & 1) << 5;      // 0,32

  const bf16_t* gA = A  + (size_t)(rowBase + aR) * K + aC;
  const bf16_t* gB = Wt + (size_t)(colBase + bR) * K + bC;

  v8f acc[2][2];
  #pragma unroll
  for (int a = 0; a < 2; ++a)
    #pragma unroll
    for (int b = 0; b < 2; ++b)
      #pragma unroll
      for (int e = 0; e < 8; ++e) acc[a][b][e] = 0.0f;

  // ---- prologue: tile 0 ----
  async_copy_b128(gA,     &As[0][aR][aC]);
  async_copy_b128(gA + 8, &As[0][aR][aC + 8]);
  #pragma unroll
  for (int j = 0; j < 4; ++j)
    async_copy_b128(gB + 8 * j, &Bs[0][bR][bC + 8 * j]);
  wait_async_zero();
  __syncthreads();

  const int ntiles = K >> 6;
  int buf = 0;
  for (int i = 0; i < ntiles; ++i) {
    const bool more = (i + 1) < ntiles;
    if (more) {
      const bf16_t* pA = gA + (size_t)(i + 1) * 64;
      const bf16_t* pB = gB + (size_t)(i + 1) * 64;
      const int nb = buf ^ 1;
      async_copy_b128(pA,     &As[nb][aR][aC]);
      async_copy_b128(pA + 8, &As[nb][aR][aC + 8]);
      #pragma unroll
      for (int j = 0; j < 4; ++j)
        async_copy_b128(pB + 8 * j, &Bs[nb][bR][bC + 8 * j]);
    }
    // ---- compute from LDS[buf]: 2 k-chunks of 32, 2x2 WMMA each ----
    #pragma unroll
    for (int c = 0; c < 2; ++c) {
      union U { v16bf v; uint4 u[2]; } afr[2], bfr[2];
      #pragma unroll
      for (int mt = 0; mt < 2; ++mt) {
        const uint4* p = (const uint4*)&As[buf][wy * 32 + mt * 16 + lr][0];
        afr[mt].u[0] = p[c * 4 + hi];         // K-local 0-7 / 8-15
        afr[mt].u[1] = p[c * 4 + hi + 2];     // K-local 16-23 / 24-31
      }
      #pragma unroll
      for (int nt = 0; nt < 2; ++nt) {
        const uint4* q = (const uint4*)&Bs[buf][wx * 32 + nt * 16 + lr][0];
        bfr[nt].u[0] = q[c * 4 + hi];         // K-local 0-7 / 8-15
        bfr[nt].u[1] = q[c * 4 + hi + 2];     // K-local 16-23 / 24-31
      }
      #pragma unroll
      for (int mt = 0; mt < 2; ++mt)
        #pragma unroll
        for (int nt = 0; nt < 2; ++nt)
          acc[mt][nt] = __builtin_amdgcn_wmma_f32_16x16x32_bf16(
              false, afr[mt].v, false, bfr[nt].v, (short)0, acc[mt][nt], false, false);
    }
    if (more) {
      wait_async_zero();
      __syncthreads();
      buf ^= 1;
    }
  }

  // ---- epilogue: C 16x16 f32 layout: col = lane&15, row = e + 8*hi ----
  #pragma unroll
  for (int nt = 0; nt < 2; ++nt) {
    const int col = colBase + wx * 32 + nt * 16 + lr;
    const float bv = bias[col];
    #pragma unroll
    for (int mt = 0; mt < 2; ++mt) {
      #pragma unroll
      for (int e = 0; e < 8; ++e) {
        const int row = rowBase + wy * 32 + mt * 16 + e + hi * 8;
        float v = acc[mt][nt][e] + bv;
        if (act == 1)      v = siluf(v);
        else if (act == 2) v = softplusf(v);
        if (resid) v += resid[(size_t)row * N + col];
        out[(size_t)row * N + col] = v;
        if (bfout && col < bfW) bfout[(size_t)row * bfW + col] = (bf16_t)v;
      }
    }
  }
}

// ---------------------------------------------------------------------------
// Selective scan. One thread per (channel, 8-state group); 8 threads/channel.
// g1: ROWS x 4096 (silu applied): [:, :2048] = xs, [:, 2048:] = silu(res)
// xdbl: ROWS x 2176 : [64:128] = B_sel, [128:] = C_sel
// ---------------------------------------------------------------------------
__global__ __launch_bounds__(256) void s6_scan(
    const float* __restrict__ g1, const float* __restrict__ xdbl,
    const float* __restrict__ delta, const float* __restrict__ A_log,
    const float* __restrict__ Dp, bf16_t* __restrict__ ybf) {
  const int t    = threadIdx.x;
  const int chan = blockIdx.x * 32 + (t >> 3);   // 0..8191
  const int sg   = t & 7;
  const int d    = chan & (D_INNER - 1);
  const int b    = chan >> 11;

  float A[8], h[8];
  #pragma unroll
  for (int j = 0; j < 8; ++j) {
    A[j] = -__expf(A_log[d * 64 + sg * 8 + j]);
    h[j] = 0.0f;
  }
  const float dpv = Dp[d];

  for (int tt = 0; tt < SEQL; ++tt) {
    const size_t r  = (size_t)b * SEQL + tt;
    const float dv  = delta[r * D_INNER + d];
    const float xv  = g1[r * 4096 + d];
    const float Cv  = xdbl[r * 2176 + 128 + d];
    const float4 b0 = *(const float4*)(xdbl + r * 2176 + 64 + sg * 8);
    const float4 b1 = *(const float4*)(xdbl + r * 2176 + 64 + sg * 8 + 4);
    const float Bv[8] = {b0.x, b0.y, b0.z, b0.w, b1.x, b1.y, b1.z, b1.w};
    const float dx = dv * xv;
    float s = 0.0f;
    #pragma unroll
    for (int j = 0; j < 8; ++j) {
      h[j] = __expf(dv * A[j]) * h[j] + dx * Bv[j];
      s += h[j];
    }
    s += __shfl_xor(s, 1);
    s += __shfl_xor(s, 2);
    s += __shfl_xor(s, 4);
    if (sg == 0) {
      const float rv = g1[r * 4096 + 2048 + d];  // silu(res), already applied
      float y = Cv * s + xv * dpv;
      y *= rv;
      ybf[r * D_INNER + d] = (bf16_t)y;
    }
  }
}

// ---------------------------------------------------------------------------
// LayerNorm over D_MODEL=1024, one block (256 threads) per row
// ---------------------------------------------------------------------------
__global__ __launch_bounds__(256) void layernorm_k(
    const float* __restrict__ z, const float* __restrict__ gamma,
    const float* __restrict__ beta, float* __restrict__ out) {
  __shared__ float redS[8], redQ[8];
  const int row = blockIdx.x;
  const int t   = threadIdx.x;
  const float4 v = ((const float4*)(z + (size_t)row * D_MODEL))[t];
  float s = v.x + v.y + v.z + v.w;
  float q = v.x * v.x + v.y * v.y + v.z * v.z + v.w * v.w;
  #pragma unroll
  for (int o = 16; o > 0; o >>= 1) {
    s += __shfl_xor(s, o);
    q += __shfl_xor(q, o);
  }
  if ((t & 31) == 0) { redS[t >> 5] = s; redQ[t >> 5] = q; }
  __syncthreads();
  s = 0.0f; q = 0.0f;
  #pragma unroll
  for (int i = 0; i < 8; ++i) { s += redS[i]; q += redQ[i]; }
  const float mu  = s * (1.0f / D_MODEL);
  const float var = q * (1.0f / D_MODEL) - mu * mu;
  const float rs  = rsqrtf(var + 1e-5f);
  const float4 g  = ((const float4*)gamma)[t];
  const float4 be = ((const float4*)beta)[t];
  float4 o;
  o.x = g.x * (v.x - mu) * rs + be.x;
  o.y = g.y * (v.y - mu) * rs + be.y;
  o.z = g.z * (v.z - mu) * rs + be.z;
  o.w = g.w * (v.w - mu) * rs + be.w;
  ((float4*)(out + (size_t)row * D_MODEL))[t] = o;
}

// ---------------------------------------------------------------------------
extern "C" void kernel_launch(void* const* d_in, const int* in_sizes, int n_in,
                              void* d_out, int out_size, void* d_ws, size_t ws_size,
                              hipStream_t stream) {
  const float* x     = (const float*)d_in[0];
  const float* Wi    = (const float*)d_in[1];
  const float* bi    = (const float*)d_in[2];
  const float* Wx    = (const float*)d_in[3];
  const float* bx    = (const float*)d_in[4];
  const float* Wdt   = (const float*)d_in[5];
  const float* bdt   = (const float*)d_in[6];
  const float* A_log = (const float*)d_in[7];
  const float* Dp    = (const float*)d_in[8];
  const float* Wo    = (const float*)d_in[9];
  const float* bo    = (const float*)d_in[10];
  const float* gamma = (const float*)d_in[11];
  const float* beta  = (const float*)d_in[12];
  float* out = (float*)d_out;

  char* ws = (char*)d_ws;
  size_t off = 0;
  auto alloc = [&](size_t bytes) -> char* {
    char* p = ws + off;
    off = (off + bytes + 255) & ~(size_t)255;
    return p;
  };

  bf16_t* xb   = (bf16_t*)alloc((size_t)ROWS * D_MODEL * 2);
  bf16_t* Wib  = (bf16_t*)alloc((size_t)D_MODEL * 4096 * 2);   // transposed 4096 x 1024
  float*  g1   = (float*) alloc((size_t)ROWS * 4096 * 4);
  bf16_t* xsb  = (bf16_t*)alloc((size_t)ROWS * D_INNER * 2);
  bf16_t* Wxb  = (bf16_t*)alloc((size_t)D_INNER * 2176 * 2);   // transposed 2176 x 2048
  float*  xdbl = (float*) alloc((size_t)ROWS * 2176 * 4);
  bf16_t* drb  = (bf16_t*)alloc((size_t)ROWS * 64 * 2);
  bf16_t* Wdtb = (bf16_t*)alloc((size_t)64 * D_INNER * 2);     // transposed 2048 x 64
  float*  dlt  = (float*) alloc((size_t)ROWS * D_INNER * 4);
  bf16_t* ybf  = (bf16_t*)alloc((size_t)ROWS * D_INNER * 2);
  bf16_t* Wob  = (bf16_t*)alloc((size_t)D_INNER * D_MODEL * 2);// transposed 1024 x 2048
  float*  zbuf = (float*) alloc((size_t)ROWS * D_MODEL * 4);

  // weight transpose-conversions (K x N fp32 -> N x K bf16)
  cvt_transpose_bf16<<<dim3(4096 / 32, D_MODEL / 32), 256, 0, stream>>>(Wi, Wib, D_MODEL, 4096);
  cvt_transpose_bf16<<<dim3(2176 / 32, D_INNER / 32), 256, 0, stream>>>(Wx, Wxb, D_INNER, 2176);
  cvt_transpose_bf16<<<dim3(D_INNER / 32, 64 / 32), 256, 0, stream>>>(Wdt, Wdtb, 64, D_INNER);
  cvt_transpose_bf16<<<dim3(D_MODEL / 32, D_INNER / 32), 256, 0, stream>>>(Wo, Wob, D_INNER, D_MODEL);

  // activation conversion: x -> bf16
  cvt_to_bf16<<<dim3(4096), dim3(256), 0, stream>>>(x, xb, (long)ROWS * D_MODEL);

  // 1) g1 = silu(x@Wi + bi) on ALL 4096 cols; fused bf16 copy of xs (cols<2048)
  gemm_bf16_wmma<<<dim3(4096 / 128, ROWS / 64), 256, 0, stream>>>(
      xb, Wib, bi, g1, nullptr, xsb, D_INNER, ROWS, 4096, D_MODEL, 1);

  // 2) xdbl = xs@Wx + bx; fused bf16 copy of d_raw (cols<64)
  gemm_bf16_wmma<<<dim3(2176 / 128, ROWS / 64), 256, 0, stream>>>(
      xsb, Wxb, bx, xdbl, nullptr, drb, 64, ROWS, 2176, D_INNER, 0);

  // 3) delta = softplus(d_raw@Wdt + bdt)
  gemm_bf16_wmma<<<dim3(D_INNER / 128, ROWS / 64), 256, 0, stream>>>(
      drb, Wdtb, bdt, dlt, nullptr, nullptr, 0, ROWS, D_INNER, 64, 2);

  // 4) selective scan (fused +xs*Dp and *silu(res)); emits y in bf16
  s6_scan<<<dim3(ROWS / 32), 256, 0, stream>>>(g1, xdbl, dlt, A_log, Dp, ybf);

  // 5) zbuf = y@Wo + bo + x (residual fused)
  gemm_bf16_wmma<<<dim3(D_MODEL / 128, ROWS / 64), 256, 0, stream>>>(
      ybf, Wob, bo, zbuf, x, nullptr, 0, ROWS, D_MODEL, D_INNER, 0);

  // 6) LayerNorm
  layernorm_k<<<dim3(ROWS), 256, 0, stream>>>(zbuf, gamma, beta, out);

  (void)in_sizes; (void)n_in; (void)out_size; (void)ws_size;
}
